// MultiHeadAttention_66374424592953
// MI455X (gfx1250) — matrix operand order repaired and checked by reading, not compiled, not run on hardware
//
#include <hip/hip_runtime.h>
#include <hip/hip_bf16.h>
#include <stdint.h>

#define DM    1024
#define NHEAD 16
#define DH    64
#define SEQ   2048
#define BS    4
#define MTOT  (BS * SEQ)   // 8192

typedef __attribute__((ext_vector_type(16))) __bf16 v16bf;
typedef __attribute__((ext_vector_type(8)))  float  v8f;
typedef unsigned short u16;
typedef int v4i_vec __attribute__((vector_size(16)));   // matches builtin param

#define AS1 __attribute__((address_space(1)))
#define AS3 __attribute__((address_space(3)))

// ---- CDNA5 async global->LDS copy (ASYNCcnt-tracked), with safe fallback ----
#if defined(__has_builtin)
#if __has_builtin(__builtin_amdgcn_global_load_async_to_lds_b128)
#define USE_ASYNC_LDS 1
#endif
#endif

__device__ __forceinline__ void cp16_g2lds(u16* l, const u16* g) {
#if defined(USE_ASYNC_LDS)
  // global_load_async_to_lds_b128: per-lane 16B, no VGPR data, ASYNCcnt
  __builtin_amdgcn_global_load_async_to_lds_b128(
      (AS1 v4i_vec*)g, (AS3 v4i_vec*)l, 0, 0);
#else
  *(uint4*)l = *(const uint4*)g;
#endif
}

__device__ __forceinline__ void wait_async0() {
#if defined(USE_ASYNC_LDS)
#if __has_builtin(__builtin_amdgcn_s_wait_asynccnt)
  __builtin_amdgcn_s_wait_asynccnt(0);
#else
  asm volatile("s_wait_asynccnt 0x0" ::: "memory");
#endif
#endif
}

union FragAB {
  v16bf bf;
  uint4 q[2];
};

__device__ __forceinline__ u16 f2bf(float f) {
  unsigned int u = __float_as_uint(f);
  u += 0x7FFFu + ((u >> 16) & 1u);   // round-to-nearest-even
  return (u16)(u >> 16);
}

__device__ __forceinline__ v8f wmma_bf16(v16bf a, v16bf b, v8f c) {
  // D = A(16x32) * B(32x16) + C, f32 accumulate
  return __builtin_amdgcn_wmma_f32_16x16x32_bf16(
      false, a, false, b, (short)0, c, false, false);
}

// ---------------- elementwise fp32 -> bf16 ----------------
__global__ __launch_bounds__(256)
void f32_to_bf16_kernel(const float* __restrict__ in, u16* __restrict__ out, int n) {
  int i = blockIdx.x * 256 + threadIdx.x;
  if (i < n) out[i] = f2bf(in[i]);
}

// ---------------- transpose 1024x1024 weight, fp32 -> bf16 ----------------
__global__ __launch_bounds__(256)
void transpose_w_kernel(const float* __restrict__ W, u16* __restrict__ Wt) {
  int i = blockIdx.x * 256 + threadIdx.x;  // 0 .. 1M-1
  int n = i >> 10;
  int k = i & 1023;
  Wt[(size_t)n * DM + k] = f2bf(W[(size_t)k * DM + n]);
}

// ---------------- WMMA GEMM:  C[M x 1024] = A[M x 1024] * Bt^T + bias ----------------
// A: bf16 row-major (M x K).  Bt: bf16 row-major (N x K) i.e. B transposed.
// mode 0: bf16 out row-major; mode 1: bf16 out transposed per-head (Vt layout
//         [b][h][dh][l]); mode 2: fp32 out row-major.
__global__ __launch_bounds__(256)
void gemm_bf16_kernel(const u16* __restrict__ A, const u16* __restrict__ Bt,
                      const float* __restrict__ bias, void* __restrict__ out,
                      int mode) {
  __shared__ __align__(16) u16 As[128 * 32];
  __shared__ __align__(16) u16 Bs[128 * 32];

  const int tid   = threadIdx.x;
  const int lane  = tid & 31;
  const int wave  = tid >> 5;       // 0..7
  const int waveM = wave & 3;       // 4 waves along M
  const int waveN = wave >> 2;      // 2 waves along N
  const int half  = lane >> 4;      // 0/1  (K-half selector)
  const int l16   = lane & 15;
  const int m0    = blockIdx.x * 128;
  const int n0    = blockIdx.y * 128;

  v8f acc[2][4];
#pragma unroll
  for (int mi = 0; mi < 2; ++mi)
#pragma unroll
    for (int ni = 0; ni < 4; ++ni) acc[mi][ni] = {};

  for (int kk = 0; kk < DM; kk += 32) {
    __syncthreads();
    // cooperative async stage of A tile (128x32) and Bt tile (128x32) into LDS
#pragma unroll
    for (int i = 0; i < 2; ++i) {
      int idx = tid + i * 256;       // 0..511
      int row = idx >> 2;
      int seg = idx & 3;
      cp16_g2lds(&As[row * 32 + seg * 8],
                 A + (size_t)(m0 + row) * DM + kk + seg * 8);
      cp16_g2lds(&Bs[row * 32 + seg * 8],
                 Bt + (size_t)(n0 + row) * DM + kk + seg * 8);
    }
    if (kk + 32 < DM) {  // prefetch next K-slab into cache
      __builtin_prefetch(A + (size_t)(m0 + (tid >> 1)) * DM + kk + 32, 0, 0);
      __builtin_prefetch(Bt + (size_t)(n0 + (tid >> 1)) * DM + kk + 32, 0, 0);
    }
    wait_async0();
    __syncthreads();

    FragAB a[2];
#pragma unroll
    for (int mi = 0; mi < 2; ++mi) {
      int row  = waveM * 32 + mi * 16 + l16;
      int base = row * 32 + half * 8;          // K = kb..kb+7, then 16+kb..
      a[mi].q[0] = *(const uint4*)&As[base];
      a[mi].q[1] = *(const uint4*)&As[base + 16];
    }
    FragAB b[4];
#pragma unroll
    for (int ni = 0; ni < 4; ++ni) {
      int nrow = waveN * 64 + ni * 16 + l16;
      int base = nrow * 32 + half * 8;         // K = kb..kb+7, then 16+kb..
      b[ni].q[0] = *(const uint4*)&Bs[base];
      b[ni].q[1] = *(const uint4*)&Bs[base + 16];
    }
#pragma unroll
    for (int mi = 0; mi < 2; ++mi)
#pragma unroll
      for (int ni = 0; ni < 4; ++ni)
        acc[mi][ni] = wmma_bf16(a[mi].bf, b[ni].bf, acc[mi][ni]);
  }

  // epilogue: bias + store
#pragma unroll
  for (int mi = 0; mi < 2; ++mi) {
#pragma unroll
    for (int ni = 0; ni < 4; ++ni) {
      int n    = n0 + waveN * 64 + ni * 16 + l16;
      float bv = bias[n];
      int mbase = m0 + waveM * 32 + mi * 16 + half * 8;
#pragma unroll
      for (int r = 0; r < 8; ++r) {
        float v = acc[mi][ni][r] + bv;
        int m = mbase + r;
        if (mode == 0) {
          ((u16*)out)[(size_t)m * DM + n] = f2bf(v);
        } else if (mode == 1) {
          int b_ = m >> 11, l_ = m & (SEQ - 1);
          int h_ = n >> 6,  dh = n & (DH - 1);
          ((u16*)out)[(((size_t)b_ * NHEAD + h_) * DH + dh) * SEQ + l_] = f2bf(v);
        } else {
          ((float*)out)[(size_t)m * DM + n] = v;
        }
      }
    }
  }
}

// ---------------- flash attention (one (b,h), 64 queries per WG) ----------------
// Q,K: bf16 [b][l][h*64+dh] (row-major, D=1024). Vt: bf16 [b][h][dh][l].
// Ctx out: bf16 [b][l][h*64+dh].
__global__ __launch_bounds__(128)
void attn_kernel(const u16* __restrict__ Q, const u16* __restrict__ K,
                 const u16* __restrict__ Vt, u16* __restrict__ Ctx) {
  __shared__ __align__(16) u16 Ks[64 * 64];        // [key][dh]
  __shared__ __align__(16) u16 Vs[64 * 64];        // [dh][key]
  __shared__ __align__(16) u16 Ps[4 * 16 * 64];    // per-wave P [q][key]

  const int tid  = threadIdx.x;
  const int lane = tid & 31;
  const int wave = tid >> 5;                 // 0..3
  const int half = lane >> 4;
  const int l16  = lane & 15;
  const int bh   = blockIdx.x;               // b*16+h
  const int b    = bh >> 4;
  const int h    = bh & 15;
  const int qw   = blockIdx.y * 64 + wave * 16;   // first query row of this wave

  // Q fragments (16 queries x 64 dh -> two 16x32 A-fragments), loaded once
  FragAB qa[2];
  {
    size_t qrow = ((size_t)b * SEQ + qw + l16) * DM + h * DH;
#pragma unroll
    for (int c = 0; c < 2; ++c) {
      size_t base = qrow + c * 32 + half * 8;
      qa[c].q[0] = *(const uint4*)(Q + base);
      qa[c].q[1] = *(const uint4*)(Q + base + 16);
    }
  }

  v8f O[4];
#pragma unroll
  for (int nt = 0; nt < 4; ++nt) O[nt] = {};
  float mrow[8], lrow[8];
#pragma unroll
  for (int r = 0; r < 8; ++r) { mrow[r] = -3.0e38f; lrow[r] = 0.0f; }

  const size_t kbase  = (size_t)b * SEQ * DM + h * DH;   // K row base
  const size_t vtbase = (size_t)bh * DH * SEQ;           // Vt base for (b,h)

  for (int kb0 = 0; kb0 < SEQ; kb0 += 64) {
    __syncthreads();
    // async stage: K tile (64 keys x 64 dh) and Vt tile (64 dh x 64 keys)
#pragma unroll
    for (int i = 0; i < 4; ++i) {
      int idx = tid + i * 128;       // 0..511
      int row = idx >> 3;
      int seg = idx & 7;
      cp16_g2lds(&Ks[row * 64 + seg * 8],
                 K + kbase + (size_t)(kb0 + row) * DM + seg * 8);
      cp16_g2lds(&Vs[row * 64 + seg * 8],
                 Vt + vtbase + (size_t)row * SEQ + kb0 + seg * 8);
    }
    wait_async0();
    __syncthreads();

    // S = Q * K^T (16 q x 64 keys) : 4 N-tiles x 2 K-chunks
    v8f S[4];
#pragma unroll
    for (int nt = 0; nt < 4; ++nt) {
      FragAB kf[2];
      int nrow = nt * 16 + l16;
#pragma unroll
      for (int kc = 0; kc < 2; ++kc) {
        int base = nrow * 64 + kc * 32 + half * 8;
        kf[kc].q[0] = *(const uint4*)&Ks[base];
        kf[kc].q[1] = *(const uint4*)&Ks[base + 16];
      }
      v8f z = {};
      S[nt] = wmma_bf16(qa[1].bf, kf[1].bf, wmma_bf16(qa[0].bf, kf[0].bf, z));
    }

    // scale + online softmax
#pragma unroll
    for (int nt = 0; nt < 4; ++nt)
#pragma unroll
      for (int r = 0; r < 8; ++r) S[nt][r] *= 0.125f;   // 1/sqrt(64)

#pragma unroll
    for (int r = 0; r < 8; ++r) {
      float v = fmaxf(fmaxf(S[0][r], S[1][r]), fmaxf(S[2][r], S[3][r]));
      v = fmaxf(v, __shfl_xor(v, 1));
      v = fmaxf(v, __shfl_xor(v, 2));
      v = fmaxf(v, __shfl_xor(v, 4));
      v = fmaxf(v, __shfl_xor(v, 8));
      float mn    = fmaxf(mrow[r], v);
      float alpha = __expf(mrow[r] - mn);
      mrow[r] = mn;
      lrow[r] *= alpha;
#pragma unroll
      for (int nt = 0; nt < 4; ++nt) O[nt][r] *= alpha;
    }

    u16* Pw = Ps + wave * (16 * 64);
    float rs[8];
#pragma unroll
    for (int r = 0; r < 8; ++r) rs[r] = 0.0f;
#pragma unroll
    for (int nt = 0; nt < 4; ++nt)
#pragma unroll
      for (int r = 0; r < 8; ++r) {
        float p = __expf(S[nt][r] - mrow[r]);
        rs[r] += p;
        Pw[(half * 8 + r) * 64 + nt * 16 + l16] = f2bf(p);  // C-layout -> LDS
      }
#pragma unroll
    for (int r = 0; r < 8; ++r) {
      float v = rs[r];
      v += __shfl_xor(v, 1);
      v += __shfl_xor(v, 2);
      v += __shfl_xor(v, 4);
      v += __shfl_xor(v, 8);
      lrow[r] += v;
    }

    __syncthreads();   // make P visible across lanes (and align waves)

    // reload P as A-fragments (16 q x 64 keys)
    FragAB pa[2];
#pragma unroll
    for (int kc = 0; kc < 2; ++kc) {
      int base = l16 * 64 + kc * 32 + half * 8;
      pa[kc].q[0] = *(const uint4*)&Pw[base];
      pa[kc].q[1] = *(const uint4*)&Pw[base + 16];
    }
    // O += P * V  (4 dh-tiles x 2 key-chunks)
#pragma unroll
    for (int nt = 0; nt < 4; ++nt) {
      FragAB vf[2];
      int nrow = nt * 16 + l16;
#pragma unroll
      for (int kc = 0; kc < 2; ++kc) {
        int base = nrow * 64 + kc * 32 + half * 8;
        vf[kc].q[0] = *(const uint4*)&Vs[base];
        vf[kc].q[1] = *(const uint4*)&Vs[base + 16];
      }
      O[nt] = wmma_bf16(pa[1].bf, vf[1].bf, wmma_bf16(pa[0].bf, vf[0].bf, O[nt]));
    }
  }

  // normalize + store context (bf16, natural [b][l][h*64+dh] layout)
#pragma unroll
  for (int nt = 0; nt < 4; ++nt)
#pragma unroll
    for (int r = 0; r < 8; ++r) {
      float v = O[nt][r] / lrow[r];
      int row = qw + half * 8 + r;
      Ctx[((size_t)b * SEQ + row) * DM + h * DH + nt * 16 + l16] = f2bf(v);
    }
}

// ---------------- host-side orchestration ----------------
extern "C" void kernel_launch(void* const* d_in, const int* in_sizes, int n_in,
                              void* d_out, int out_size, void* d_ws, size_t ws_size,
                              hipStream_t stream) {
  const float* X  = (const float*)d_in[0];
  const float* Wq = (const float*)d_in[1];
  const float* bq = (const float*)d_in[2];
  const float* Wk = (const float*)d_in[3];
  const float* bk = (const float*)d_in[4];
  const float* Wv = (const float*)d_in[5];
  const float* bv = (const float*)d_in[6];
  const float* Wo = (const float*)d_in[7];
  const float* bo = (const float*)d_in[8];

  // workspace layout (ushort elements)
  u16* ws   = (u16*)d_ws;
  u16* Xbf  = ws;                      // 8192*1024
  u16* Wqt  = Xbf + (size_t)MTOT * DM; // 1024*1024 each
  u16* Wkt  = Wqt + (size_t)DM * DM;
  u16* Wvt  = Wkt + (size_t)DM * DM;
  u16* Wot  = Wvt + (size_t)DM * DM;
  u16* Qbf  = Wot + (size_t)DM * DM;
  u16* Kbf  = Qbf + (size_t)MTOT * DM;
  u16* Vtw  = Kbf + (size_t)MTOT * DM; // transposed per-head [b][h][dh][l]
  u16* Ctx  = Vtw + (size_t)MTOT * DM;

  const int nX = MTOT * DM;            // 8388608
  f32_to_bf16_kernel<<<nX / 256, 256, 0, stream>>>(X, Xbf, nX);
  transpose_w_kernel<<<(DM * DM) / 256, 256, 0, stream>>>(Wq, Wqt);
  transpose_w_kernel<<<(DM * DM) / 256, 256, 0, stream>>>(Wk, Wkt);
  transpose_w_kernel<<<(DM * DM) / 256, 256, 0, stream>>>(Wv, Wvt);
  transpose_w_kernel<<<(DM * DM) / 256, 256, 0, stream>>>(Wo, Wot);

  dim3 gg(MTOT / 128, DM / 128);       // (64, 8)
  gemm_bf16_kernel<<<gg, 256, 0, stream>>>(Xbf, Wqt, bq, Qbf, 0);
  gemm_bf16_kernel<<<gg, 256, 0, stream>>>(Xbf, Wkt, bk, Kbf, 0);
  gemm_bf16_kernel<<<gg, 256, 0, stream>>>(Xbf, Wvt, bv, Vtw, 1);

  attn_kernel<<<dim3(BS * NHEAD, SEQ / 64), 128, 0, stream>>>(Qbf, Kbf, Vtw, Ctx);

  gemm_bf16_kernel<<<gg, 256, 0, stream>>>(Ctx, Wot, bo, d_out, 2);
}
